// CrossModalCodebook_10204842295876
// MI455X (gfx1250) — matrix operand order repaired and checked by reading, not compiled, not run on hardware
//
#include <hip/hip_runtime.h>
#include <hip/hip_bf16.h>

#define DEV __device__ __forceinline__

typedef __attribute__((ext_vector_type(8)))  __bf16 v8bf;
typedef __attribute__((ext_vector_type(16))) __bf16 v16bf;
typedef __attribute__((ext_vector_type(8)))  float  v8f;
typedef int v4i_ __attribute__((vector_size(16)));  // matches builtin param type

static constexpr int B_  = 32768;
static constexpr int DA_ = 1024;
static constexpr int DB_ = 768;
static constexpr int CD_ = 128;
static constexpr int NC_ = 8192;
static constexpr int H_  = 512;

// ---------------------------------------------------------------------------
// gfx1250 async global->LDS copy (ASYNCcnt-tracked). Fall back to plain
// load + ds_store if this toolchain doesn't declare the builtins.
// Builtin signature (from probe diagnostics): first param is
//   int __vector(4) addrspace(1)*  -- cast accordingly.
// ---------------------------------------------------------------------------
#if __has_builtin(__builtin_amdgcn_global_load_async_to_lds_b128) && \
    __has_builtin(__builtin_amdgcn_s_wait_asynccnt)
#define USE_ASYNC_LDS 1
#else
#define USE_ASYNC_LDS 0
#endif

DEV void copy16_g2l(const __bf16* g, __bf16* l) {
#if USE_ASYNC_LDS
  __builtin_amdgcn_global_load_async_to_lds_b128(
      (__attribute__((address_space(1))) v4i_*)g,
      (__attribute__((address_space(3))) v4i_*)l, 0, 0);
#else
  *reinterpret_cast<v8bf*>(l) = *reinterpret_cast<const v8bf*>(g);
#endif
}

DEV void stage_fence() {
#if USE_ASYNC_LDS
  __builtin_amdgcn_s_wait_asynccnt(0);
#endif
  __syncthreads();
}

// ---------------------------------------------------------------------------
// WMMA fragment loaders (bf16, 16x16x32), per CDNA5 ISA 7.12.2 layouts.
// A-matrix 16x32 (MxK): lanes 0-15 = M, VGPR0-3 hold K 0-7 (lo) / 8-15 (hi),
//                       VGPR4-7 hold K 16-23 (lo) / 24-31 (hi).
// B-matrix 32x16 (KxN): lanes = N; same K order as A (K 0-7/16-23 lo, 8-15/24-31 hi).
// ---------------------------------------------------------------------------
DEV v16bf load_fragA(const __bf16* __restrict__ p, int ldk) {
  const int lane = threadIdx.x & 31;
  const int row  = lane & 15;
  const int kb   = (lane & 16) ? 8 : 0;
  const __bf16* q = p + (size_t)row * ldk + kb;
  v8bf c0 = *reinterpret_cast<const v8bf*>(q);
  v8bf c1 = *reinterpret_cast<const v8bf*>(q + 16);
  v16bf f;
#pragma unroll
  for (int i = 0; i < 8; ++i) { f[i] = c0[i]; f[8 + i] = c1[i]; }
  return f;
}

// Works for global or LDS (generic) pointers; compiler infers ds_load for LDS.
DEV v16bf load_fragB(const __bf16* p, int ldk) {
  const int lane = threadIdx.x & 31;
  const int n    = lane & 15;
  const int kh   = (lane & 16) ? 8 : 0;
  const __bf16* q = p + (size_t)n * ldk + kh;
  v8bf c0 = *reinterpret_cast<const v8bf*>(q);
  v8bf c1 = *reinterpret_cast<const v8bf*>(q + 16);
  v16bf f;
#pragma unroll
  for (int i = 0; i < 8; ++i) { f[i] = c0[i]; f[8 + i] = c1[i]; }
  return f;
}

DEV v8f wmma_bf16(v16bf a, v16bf b, v8f c) {
  return __builtin_amdgcn_wmma_f32_16x16x32_bf16(
      /*neg_a=*/false, a, /*neg_b=*/false, b,
      /*c_mod=*/(short)0, c, /*reuse_a=*/false, /*reuse_b=*/false);
}

// ---------------------------------------------------------------------------
// GEMM: C[M,N] = A[M,K](bf16) * W[N,K]^T(bf16) + bias, optional ReLU.
// Wave tile 32x64 (2x4 WMMA tiles), 8 waves -> 128x128 workgroup tile.
// W K-slice (128x32) is staged per K-step into double-buffered LDS (shared by
// all 8 waves, 4x fewer vector-memory requests for W). Row stride padded to
// 40 bf16 so the 16-lane ds_load_b128 fragment reads are bank-conflict-free.
// grid = (M/128, N/128); M,N,K multiples of (128,128,32).
// ---------------------------------------------------------------------------
template<int RELU, int WF32, int WBF16>
__global__ __launch_bounds__(256)
void gemm_bf16_k(const __bf16* __restrict__ A, const __bf16* __restrict__ W,
                 const float* __restrict__ bias,
                 float* __restrict__ Cf, __bf16* __restrict__ Cb,
                 int N, int K) {
  constexpr int WM = 2, WN = 4;
  constexpr int LDW = 40;  // padded LDS row stride, bf16 elems (32 data + 8 pad)
  __shared__ alignas(16) __bf16 sW[2][128 * LDW];

  const int tid  = threadIdx.x;
  const int wave = tid >> 5;
  const int lane = tid & 31;
  const int wm = wave & 3;
  const int wn = wave >> 2;
  const int m0  = blockIdx.x * 128 + wm * (WM * 16);
  const int n0g = blockIdx.y * 128;        // WG's base row of W
  const int n0l = wn * (WN * 16);          // wave's base row within LDS tile

  // staging map: thread t stages 16 elems of W row (t>>1) at col (t&1)*16
  const int srow = tid >> 1;
  const int scol = (tid & 1) * 16;

  auto stage = [&](int buf, int k) {
    const __bf16* src = W + (size_t)(n0g + srow) * K + k + scol;
    __bf16* dst = &sW[buf][srow * LDW + scol];
    copy16_g2l(src,     dst);
    copy16_g2l(src + 8, dst + 8);
  };

  v8f acc[WM][WN] = {};

  const int nk = K / 32;
  int cur = 0;
  stage(0, 0);
  stage_fence();
  for (int kt = 0; kt < nk; ++kt) {
    const int k = kt * 32;
    if (kt + 1 < nk) stage(cur ^ 1, k + 32);

    v16bf a[WM], b[WN];
#pragma unroll
    for (int i = 0; i < WM; ++i)
      a[i] = load_fragA(A + (size_t)(m0 + 16 * i) * K + k, K);
#pragma unroll
    for (int j = 0; j < WN; ++j)
      b[j] = load_fragB(&sW[cur][(n0l + 16 * j) * LDW], LDW);
#pragma unroll
    for (int i = 0; i < WM; ++i)
#pragma unroll
      for (int j = 0; j < WN; ++j)
        acc[i][j] = wmma_bf16(a[i], b[j], acc[i][j]);

    if (kt + 1 < nk) { stage_fence(); cur ^= 1; }
  }

  // C layout: VGPR r, lanes 0-15 -> (row r, col lane), lanes 16-31 -> (row r+8).
  const int cc = lane & 15;
  const int rr = (lane >> 4) << 3;
#pragma unroll
  for (int j = 0; j < WN; ++j) {
    const int col = n0g + n0l + 16 * j + cc;
    const float bv = bias ? bias[col] : 0.f;
#pragma unroll
    for (int i = 0; i < WM; ++i) {
#pragma unroll
      for (int r = 0; r < 8; ++r) {
        const int row = m0 + 16 * i + rr + r;
        float v = acc[i][j][r] + bv;
        if (RELU) v = fmaxf(v, 0.f);
        const size_t off = (size_t)row * N + col;
        if (WF32)  Cf[off] = v;
        if (WBF16) Cb[off] = (__bf16)v;
      }
    }
  }
}

// ---------------------------------------------------------------------------
// Row LayerNorm (biased variance, eps=1e-5), optional ReLU; emits fp32 and/or
// bf16 copies. One 128-thread block per row.
// ---------------------------------------------------------------------------
template<int RELU, int WF32, int WBF16>
__global__ __launch_bounds__(128)
void layernorm_k(const float* __restrict__ X, const float* __restrict__ g,
                 const float* __restrict__ be,
                 float* __restrict__ outF, __bf16* __restrict__ outB, int N) {
  __shared__ float sh[8];
  const int row  = blockIdx.x;
  const float* x = X + (size_t)row * N;
  const int lane = threadIdx.x & 31;
  const int wv   = threadIdx.x >> 5;
  float s = 0.f, s2 = 0.f;
  for (int i = threadIdx.x; i < N; i += 128) { float v = x[i]; s += v; s2 += v * v; }
#pragma unroll
  for (int off = 16; off >= 1; off >>= 1) {
    s  += __shfl_xor(s,  off, 32);
    s2 += __shfl_xor(s2, off, 32);
  }
  if (lane == 0) { sh[wv] = s; sh[4 + wv] = s2; }
  __syncthreads();
  if (threadIdx.x == 0) {
    float ts = sh[0] + sh[1] + sh[2] + sh[3];
    float t2 = sh[4] + sh[5] + sh[6] + sh[7];
    float m   = ts / (float)N;
    float var = t2 / (float)N - m * m;
    sh[0] = m;
    sh[1] = rsqrtf(var + 1e-5f);
  }
  __syncthreads();
  const float m = sh[0], inv = sh[1];
  for (int i = threadIdx.x; i < N; i += 128) {
    float y = (x[i] - m) * inv * g[i] + be[i];
    if (RELU) y = fmaxf(y, 0.f);
    const size_t off = (size_t)row * N + i;
    if (WF32)  outF[off] = y;
    if (WBF16) outB[off] = (__bf16)y;
  }
}

// ---------------------------------------------------------------------------
// Fused VQ: for 32 rows/wave, sweep all NC codes with WMMA (z . E^T), track
// running min of (|e|^2 - 2 z.e)  (|z|^2 is row-constant -> argmin-invariant),
// cross-lane reduce per row, gather q = E[idx], accumulate commitment loss.
// Codebook is staged through double-buffered LDS in 64-row blocks (one fill
// serves all 8 waves instead of 8 private streams of the full 2 MB codebook).
// Row stride padded to 136 bf16 for conflict-free ds_load_b128 fragments.
// ---------------------------------------------------------------------------
__global__ __launch_bounds__(256)
void vq_argmin_k(const __bf16* __restrict__ Zb, const float* __restrict__ Zf,
                 const __bf16* __restrict__ Eb, const float* __restrict__ Ef,
                 const float* __restrict__ enorm,
                 float* __restrict__ idx_out, float* __restrict__ q_out,
                 __bf16* __restrict__ qb_out, float* __restrict__ comm_acc) {
  constexpr int LDE = 136;  // padded LDS row stride (128 data + 8 pad)
  __shared__ alignas(16) __bf16 sE[2][64 * LDE];

  const int tid  = threadIdx.x;
  const int lane = tid & 31;
  const int wave = tid >> 5;
  const int m0   = (blockIdx.x * 8 + wave) * 32;

  // staging map: thread t stages 32 elems of E row (t>>2) at col (t&3)*32
  const int srow = tid >> 2;
  const int scol = (tid & 3) * 32;
  auto stageE = [&](int buf, int nrow0) {
    const __bf16* src = Eb + (size_t)(nrow0 + srow) * CD_ + scol;
    __bf16* dst = &sE[buf][srow * LDE + scol];
#pragma unroll
    for (int i = 0; i < 4; ++i) copy16_g2l(src + 8 * i, dst + 8 * i);
  };

  v16bf za[2][4];
#pragma unroll
  for (int mt = 0; mt < 2; ++mt)
#pragma unroll
    for (int kk = 0; kk < 4; ++kk)
      za[mt][kk] = load_fragA(Zb + (size_t)(m0 + 16 * mt) * CD_ + 32 * kk, CD_);

  float best[2][8];
  int   bidx[2][8];
#pragma unroll
  for (int mt = 0; mt < 2; ++mt)
#pragma unroll
    for (int r = 0; r < 8; ++r) { best[mt][r] = 3.402823466e38f; bidx[mt][r] = 0; }

  const int cc = lane & 15;
  constexpr int NB = NC_ / 64;
  int cur = 0;
  stageE(0, 0);
  stage_fence();
  for (int nb = 0; nb < NB; ++nb) {
    if (nb + 1 < NB) stageE(cur ^ 1, (nb + 1) * 64);
#pragma unroll
    for (int s = 0; s < 4; ++s) {
      const int nt = nb * 4 + s;
      v16bf eb[4];
#pragma unroll
      for (int kk = 0; kk < 4; ++kk)
        eb[kk] = load_fragB(&sE[cur][(s * 16) * LDE + 32 * kk], LDE);
      const float en = enorm[nt * 16 + cc];
#pragma unroll
      for (int mt = 0; mt < 2; ++mt) {
        v8f acc = {};
#pragma unroll
        for (int kk = 0; kk < 4; ++kk) acc = wmma_bf16(za[mt][kk], eb[kk], acc);
#pragma unroll
        for (int r = 0; r < 8; ++r) {
          const float sc = en - 2.f * acc[r];
          if (sc < best[mt][r]) { best[mt][r] = sc; bidx[mt][r] = nt * 16 + cc; }
        }
      }
    }
    if (nb + 1 < NB) { stage_fence(); cur ^= 1; }
  }

  float csum = 0.f;
#pragma unroll
  for (int mt = 0; mt < 2; ++mt) {
#pragma unroll
    for (int r = 0; r < 8; ++r) {
      float bv = best[mt][r];
      int   bi = bidx[mt][r];
#pragma unroll
      for (int off = 8; off >= 1; off >>= 1) {  // reduce within each 16-lane half
        const float ov = __shfl_xor(bv, off, 32);
        const int   oi = __shfl_xor(bi, off, 32);
        if (ov < bv || (ov == bv && oi < bi)) { bv = ov; bi = oi; }
      }
      const int row = m0 + 16 * mt + ((lane >> 4) << 3) + r;
      if (cc == 0) idx_out[row] = (float)bi;
      const int c0 = cc * 8;  // 16 lanes x 8 floats = 128-wide row copy
#pragma unroll
      for (int j = 0; j < 8; ++j) {
        const float e = Ef[(size_t)bi * CD_ + c0 + j];
        const size_t po = (size_t)row * CD_ + c0 + j;
        q_out[po]  = e;
        qb_out[po] = (__bf16)e;
        const float d = Zf[po] - e;
        csum += d * d;
      }
    }
  }
#pragma unroll
  for (int off = 16; off >= 1; off >>= 1) csum += __shfl_xor(csum, off, 32);
  if (lane == 0) atomicAdd(comm_acc, csum * (1.f / ((float)B_ * (float)CD_)));
}

// ---------------------------------------------------------------------------
// Small utility kernels
// ---------------------------------------------------------------------------
__global__ void cvt_bf16_k(const float* __restrict__ in, __bf16* __restrict__ out, int n) {
  const int i = blockIdx.x * 256 + threadIdx.x;
  if (i < n) out[i] = (__bf16)in[i];
}

__global__ void enorm_k(const float* __restrict__ E, float* __restrict__ en) {
  const int r = blockIdx.x * 256 + threadIdx.x;
  if (r < NC_) {
    const float* p = E + (size_t)r * CD_;
    float s = 0.f;
#pragma unroll 4
    for (int c = 0; c < CD_; ++c) { const float v = p[c]; s += v * v; }
    en[r] = s;
  }
}

__global__ void zero2_k(float* a, float* b) {
  if (threadIdx.x == 0 && blockIdx.x == 0) { *a = 0.f; *b = 0.f; }
}

// ---------------------------------------------------------------------------
// Launch
// ---------------------------------------------------------------------------
extern "C" void kernel_launch(void* const* d_in, const int* in_sizes, int n_in,
                              void* d_out, int out_size, void* d_ws, size_t ws_size,
                              hipStream_t stream) {
  (void)in_sizes; (void)n_in; (void)out_size; (void)ws_size;

  const float* x_a   = (const float*)d_in[0];
  const float* x_b   = (const float*)d_in[1];
  const float* a_w1  = (const float*)d_in[2];
  const float* a_b1  = (const float*)d_in[3];
  const float* a_g1  = (const float*)d_in[4];
  const float* a_be1 = (const float*)d_in[5];
  const float* a_w2  = (const float*)d_in[6];
  const float* a_b2  = (const float*)d_in[7];
  const float* a_g2  = (const float*)d_in[8];
  const float* a_be2 = (const float*)d_in[9];
  const float* a_dw1 = (const float*)d_in[10];
  const float* a_db1 = (const float*)d_in[11];
  const float* a_dw2 = (const float*)d_in[12];
  const float* a_db2 = (const float*)d_in[13];
  const float* b_w1  = (const float*)d_in[14];
  const float* b_b1  = (const float*)d_in[15];
  const float* b_g1  = (const float*)d_in[16];
  const float* b_be1 = (const float*)d_in[17];
  const float* b_w2  = (const float*)d_in[18];
  const float* b_b2  = (const float*)d_in[19];
  const float* b_g2  = (const float*)d_in[20];
  const float* b_be2 = (const float*)d_in[21];
  const float* b_dw1 = (const float*)d_in[22];
  const float* b_db1 = (const float*)d_in[23];
  const float* b_dw2 = (const float*)d_in[24];
  const float* b_db2 = (const float*)d_in[25];
  const float* E     = (const float*)d_in[26];

  // d_out layout: z_a, z_b, q_a, q_b, idx_a, idx_b, recon_a, recon_b, comm_a, comm_b
  float* out   = (float*)d_out;
  float* zA    = out;
  float* zB    = out + (size_t)B_ * CD_;
  float* qA    = out + 2ull * B_ * CD_;
  float* qB    = out + 3ull * B_ * CD_;
  float* idxA  = out + 4ull * B_ * CD_;
  float* idxB  = idxA + B_;
  float* recA  = idxB + B_;
  float* recB  = recA + (size_t)B_ * DA_;
  float* commA = recB + (size_t)B_ * DB_;
  float* commB = commA + 1;

  // workspace carve-up (256B aligned)
  char* ws = (char*)d_ws;
  size_t o = 0;
  auto bump = [&](size_t elems, size_t esz) {
    size_t r = o; o += ((elems * esz + 255) & ~(size_t)255); return r;
  };
  __bf16* xbf    = (__bf16*)(ws + bump((size_t)B_ * DA_, 2));
  __bf16* aw1bf  = (__bf16*)(ws + bump((size_t)H_ * DA_, 2));
  __bf16* aw2bf  = (__bf16*)(ws + bump((size_t)CD_ * H_, 2));
  __bf16* adw1bf = (__bf16*)(ws + bump((size_t)H_ * CD_, 2));
  __bf16* adw2bf = (__bf16*)(ws + bump((size_t)DA_ * H_, 2));
  __bf16* bw1bf  = (__bf16*)(ws + bump((size_t)H_ * DB_, 2));
  __bf16* bw2bf  = (__bf16*)(ws + bump((size_t)CD_ * H_, 2));
  __bf16* bdw1bf = (__bf16*)(ws + bump((size_t)H_ * CD_, 2));
  __bf16* bdw2bf = (__bf16*)(ws + bump((size_t)DB_ * H_, 2));
  __bf16* Ebf    = (__bf16*)(ws + bump((size_t)NC_ * CD_, 2));
  float*  enorm  = (float*) (ws + bump((size_t)NC_, 4));
  float*  h1     = (float*) (ws + bump((size_t)B_ * H_, 4));
  __bf16* h1n    = (__bf16*)(ws + bump((size_t)B_ * H_, 2));
  float*  h2     = (float*) (ws + bump((size_t)B_ * CD_, 4));
  __bf16* zbf    = (__bf16*)(ws + bump((size_t)B_ * CD_, 2));
  __bf16* qbf    = (__bf16*)(ws + bump((size_t)B_ * CD_, 2));
  __bf16* dech   = (__bf16*)(ws + bump((size_t)B_ * H_, 2));

  auto cvt = [&](const float* src, __bf16* dst, size_t n) {
    cvt_bf16_k<<<dim3((unsigned)((n + 255) / 256)), dim3(256), 0, stream>>>(src, dst, (int)n);
  };

  // one-time (per call) weight / codebook prep
  cvt(a_w1,  aw1bf,  (size_t)H_ * DA_);
  cvt(a_w2,  aw2bf,  (size_t)CD_ * H_);
  cvt(a_dw1, adw1bf, (size_t)H_ * CD_);
  cvt(a_dw2, adw2bf, (size_t)DA_ * H_);
  cvt(b_w1,  bw1bf,  (size_t)H_ * DB_);
  cvt(b_w2,  bw2bf,  (size_t)CD_ * H_);
  cvt(b_dw1, bdw1bf, (size_t)H_ * CD_);
  cvt(b_dw2, bdw2bf, (size_t)DB_ * H_);
  cvt(E,     Ebf,    (size_t)NC_ * CD_);
  enorm_k<<<dim3(NC_ / 256), dim3(256), 0, stream>>>(E, enorm);
  zero2_k<<<dim3(1), dim3(64), 0, stream>>>(commA, commB);

  auto run_pipeline = [&](const float* x, int Din,
                          __bf16* w1, const float* b1, const float* g1, const float* be1,
                          __bf16* w2, const float* b2, const float* g2, const float* be2,
                          __bf16* dw1, const float* db1, __bf16* dw2, const float* db2,
                          float* zOut, float* qOut, float* idxOut, float* recOut,
                          float* commOut) {
    cvt(x, xbf, (size_t)B_ * Din);
    // enc1: [B,Din] x [H,Din]^T + b1 -> h1 (f32)
    gemm_bf16_k<0, 1, 0><<<dim3(B_ / 128, H_ / 128), dim3(256), 0, stream>>>(
        xbf, w1, b1, h1, (__bf16*)nullptr, H_, Din);
    // LN + ReLU -> h1n (bf16)
    layernorm_k<1, 0, 1><<<dim3(B_), dim3(128), 0, stream>>>(
        h1, g1, be1, (float*)nullptr, h1n, H_);
    // enc2: [B,H] x [CD,H]^T + b2 -> h2 (f32)
    gemm_bf16_k<0, 1, 0><<<dim3(B_ / 128, CD_ / 128), dim3(256), 0, stream>>>(
        h1n, w2, b2, h2, (__bf16*)nullptr, CD_, H_);
    // LN -> z (f32 to d_out) + zbf (bf16)
    layernorm_k<0, 1, 1><<<dim3(B_), dim3(128), 0, stream>>>(h2, g2, be2, zOut, zbf, CD_);
    // fused VQ: argmin + gather + commitment loss
    vq_argmin_k<<<dim3(B_ / 256), dim3(256), 0, stream>>>(
        zbf, zOut, Ebf, E, enorm, idxOut, qOut, qbf, commOut);
    // dec1: [B,CD] x [H,CD]^T + db1, ReLU -> dech (bf16)
    gemm_bf16_k<1, 0, 1><<<dim3(B_ / 128, H_ / 128), dim3(256), 0, stream>>>(
        qbf, dw1, db1, (float*)nullptr, dech, H_, CD_);
    // dec2: [B,H] x [Din,H]^T + db2 -> recon (f32 to d_out)
    gemm_bf16_k<0, 1, 0><<<dim3(B_ / 128, Din / 128), dim3(256), 0, stream>>>(
        dech, dw2, db2, recOut, (__bf16*)nullptr, Din, H_);
  };

  run_pipeline(x_a, DA_, aw1bf, a_b1, a_g1, a_be1, aw2bf, a_b2, a_g2, a_be2,
               adw1bf, a_db1, adw2bf, a_db2, zA, qA, idxA, recA, commA);
  run_pipeline(x_b, DB_, bw1bf, b_b1, b_g1, b_be1, bw2bf, b_b2, b_g2, b_be2,
               bdw1bf, b_db1, bdw2bf, b_db2, zB, qB, idxB, recB, commB);
}
